// PointTextFusion_28535762715021
// MI455X (gfx1250) — compile-verified, matrix operand order checked by reading, not run on hardware
//
#include <hip/hip_runtime.h>

// ---------------------------------------------------------------------------
// PointTextFusion on gfx1250 (MI455X), wave32.
// GEMMs: v_wmma_f32_16x16x32_f16 with f16 weights pre-converted/transposed.
// W tiles double-buffered in LDS and staged with global_load_async_to_lds_b128
// (ASYNCcnt path, in-order completion => s_wait_asynccnt 4 pipelining).
// ---------------------------------------------------------------------------

#define Hc   256
#define NHc  8
#define DHc  32
#define Sc   1024
#define EPSc 1e-5f
#define Bc   8
#define NPc  16384
#define LTc  64

typedef long long i64;
typedef __attribute__((ext_vector_type(16))) _Float16 v16h;
typedef __attribute__((ext_vector_type(8)))  float    v8f;
typedef __attribute__((ext_vector_type(2)))  _Float16 h2;

// Async global->LDS 128-bit copy (CDNA5, tracked by ASYNCcnt).
// lds_off: raw 32-bit LDS byte address (low 32 bits of generic shared ptr).
__device__ __forceinline__ void async_g2l_b128(unsigned lds_off, const void* gaddr)
{
    asm volatile("global_load_async_to_lds_b128 %0, %1, off"
                 :: "v"(lds_off), "v"(gaddr) : "memory");
}
// Allow the 4 just-issued chunk copies (next tile) to stay in flight; since
// async loads complete in order, ASYNCcnt<=4 proves the previous tile landed.
__device__ __forceinline__ void wait_async_le4()
{
    asm volatile("s_wait_asynccnt 0x4" ::: "memory");
}
__device__ __forceinline__ void wait_async_le0()
{
    asm volatile("s_wait_asynccnt 0x0" ::: "memory");
}

// ---------------------------------------------------------------------------
// Weight pre-pass: W fp32 [Din][256] -> WT f16 [256][Din]  (n-major, so a
// k-tile of 32 is 64 contiguous bytes per output column).
// ---------------------------------------------------------------------------
__global__ __launch_bounds__(256) void k_wconv(
    const float* __restrict__ W, _Float16* __restrict__ WT, int Din)
{
    int idx = blockIdx.x * 256 + threadIdx.x;     // over Din*256
    if (idx >= Din * 256) return;
    int k = idx >> 8;
    int n = idx & 255;
    WT[n * Din + k] = (_Float16)W[idx];
}

// ---------------------------------------------------------------------------
// Fused GEMM: out[row, 0:256] = act( LN( A @ W + bias ) )
//   A = concat(A1 [rows x Din1], A2 [rows x Din2]) (A2 optional), fp32.
//   WT = f16 weights, [256][Din] n-major. bias/gamma/beta fp32 [256].
// Block = 64 rows x 256 cols, 256 threads (8 waves).
// Wave w owns cols [w*32, w*32+32) (2 col tiles) x 4 row tiles
//   => 8 v_wmma per k-step per wave.
// K-loop: 32-wide steps; W tile (256x32 f16 = 16KB) double-buffered and
// staged with async copies that overlap the WMMA of the previous tile.
// ---------------------------------------------------------------------------
template <int ROWS, int DMAX>
__global__ __launch_bounds__(256) void gemm256(
    const float* __restrict__ A1, int Din1,
    const float* __restrict__ A2, int Din2,
    const _Float16* __restrict__ WT,
    const float* __restrict__ bias,
    const float* __restrict__ gamma, const float* __restrict__ beta,
    float* __restrict__ out, int do_ln, int do_relu)
{
    constexpr int RT  = ROWS / 16;           // row tiles per block
    constexpr int TPR = 256 / ROWS;          // epilogue threads per row
    constexpr int ABYTES = ROWS * DMAX * 2;
    constexpr int YBYTES = ROWS * Hc * 4;
    constexpr int UBYTES = (ABYTES > YBYTES) ? ABYTES : YBYTES;
    constexpr int WELE   = 256 * 32;         // elements per W tile buffer

    __shared__ __align__(16) char     usm[UBYTES];      // A tile (f16) then Y (f32)
    __shared__ __align__(16) _Float16 ldsW[2 * WELE];   // double-buffered W tile [n][kk]
    __shared__ float redS[ROWS][TPR];
    __shared__ float redQ[ROWS][TPR];
    __shared__ float rowM[ROWS];
    __shared__ float rowR[ROWS];

    _Float16* ldsA = (_Float16*)usm;
    float*    ldsY = (float*)usm;

    const int tid  = threadIdx.x;
    const int wave = tid >> 5;
    const int lane = tid & 31;
    const int half = lane >> 4;
    const int l16  = lane & 15;
    const i64 row0 = (i64)blockIdx.x * ROWS;
    const int Din  = Din1 + Din2;
    const int nk   = Din >> 5;               // k-steps of 32

    // Per-thread async-copy bases: thread `tid` owns output column n=tid.
    const char*    wgbase = (const char*)(WT + (size_t)tid * Din);
    const unsigned wlbase = (unsigned)(size_t)&ldsW[tid * 32];

    // Prologue: kick off W tile 0 into buffer 0; overlaps A staging below.
    #pragma unroll
    for (int p = 0; p < 4; ++p)
        async_g2l_b128(wlbase + p * 16, wgbase + p * 16);

    // ---- Stage A tile (fp32 -> f16), coalesced, no divisions. ----
    for (int r = wave; r < ROWS; r += 8) {
        const i64 gr = row0 + r;
        for (int c = lane; c < Din; c += 32) {
            float v = (c < Din1) ? A1[gr * Din1 + c]
                                 : A2[gr * Din2 + (c - Din1)];
            ldsA[r * Din + c] = (_Float16)v;
        }
    }

    v8f acc[RT][2];
    #pragma unroll
    for (int rt = 0; rt < RT; ++rt) {
        acc[rt][0] = v8f{0.f,0.f,0.f,0.f,0.f,0.f,0.f,0.f};
        acc[rt][1] = v8f{0.f,0.f,0.f,0.f,0.f,0.f,0.f,0.f};
    }

    for (int ki = 0; ki < nk; ++ki) {
        __syncthreads();   // A staged (ki==0); reads of target buffer finished
        if (ki + 1 < nk) {
            // Issue next W tile into the other buffer, then require only the
            // *previous* tile to have landed (in-order completion).
            const char*    g  = wgbase + (size_t)(ki + 1) * 64;   // 32 k * 2B
            const unsigned lb = wlbase + (unsigned)(((ki + 1) & 1) ? WELE * 2 : 0);
            #pragma unroll
            for (int p = 0; p < 4; ++p)
                async_g2l_b128(lb + p * 16, g + p * 16);
            wait_async_le4();
        } else {
            wait_async_le0();
        }
        __syncthreads();   // all waves' chunks of tile ki visible in LDS

        const _Float16* wbuf = &ldsW[(ki & 1) ? WELE : 0];
        const int k0 = ki * 32;

        // B fragments for this wave's two 16-col tiles (lane = N).
        v16h bf0, bf1;
        #pragma unroll
        for (int v = 0; v < 8; ++v) {
            int bk = (v < 4) ? (2 * v + half * 8) : (16 + 2 * (v - 4) + half * 8);
            h2 t0 = *(const h2*)&wbuf[(wave * 32 + l16)      * 32 + bk];
            h2 t1 = *(const h2*)&wbuf[(wave * 32 + 16 + l16) * 32 + bk];
            bf0[2 * v] = t0[0]; bf0[2 * v + 1] = t0[1];
            bf1[2 * v] = t1[0]; bf1[2 * v + 1] = t1[1];
        }
        // A fragment per row tile (lane = M, halves split K) -> 8 WMMA.
        #pragma unroll
        for (int rt = 0; rt < RT; ++rt) {
            v16h a;
            #pragma unroll
            for (int v = 0; v < 8; ++v) {
                int bk = (v < 4) ? (2 * v + half * 8) : (16 + 2 * (v - 4) + half * 8);
                h2 t = *(const h2*)&ldsA[(rt * 16 + l16) * Din + k0 + bk];
                a[2 * v] = t[0]; a[2 * v + 1] = t[1];
            }
            acc[rt][0] = __builtin_amdgcn_wmma_f32_16x16x32_f16(
                false, a, false, bf0, (short)0, acc[rt][0], false, false);
            acc[rt][1] = __builtin_amdgcn_wmma_f32_16x16x32_f16(
                false, a, false, bf1, (short)0, acc[rt][1], false, false);
        }
    }
    __syncthreads();                   // k-loop done; usm can become Y

    // ---- Epilogue: C/D layout (lane=N, VGPR v -> row v+8*half) + bias. ----
    const int n0 = wave * 32 + l16;
    const int n1 = n0 + 16;
    #pragma unroll
    for (int rt = 0; rt < RT; ++rt) {
        #pragma unroll
        for (int v = 0; v < 8; ++v) {
            int r = rt * 16 + v + half * 8;
            ldsY[r * Hc + n0] = acc[rt][0][v] + bias[n0];
            ldsY[r * Hc + n1] = acc[rt][1][v] + bias[n1];
        }
    }
    __syncthreads();

    const int r = tid / TPR, g = tid % TPR;
    if (do_ln) {
        float s = 0.f, q = 0.f;
        for (int c = g; c < Hc; c += TPR) { float y = ldsY[r * Hc + c]; s += y; q += y * y; }
        redS[r][g] = s; redQ[r][g] = q;
        __syncthreads();
        if (g == 0) {
            float S = 0.f, Q = 0.f;
            #pragma unroll
            for (int j = 0; j < TPR; ++j) { S += redS[r][j]; Q += redQ[r][j]; }
            float mean = S * (1.f / Hc);
            float var  = Q * (1.f / Hc) - mean * mean;
            rowM[r] = mean;
            rowR[r] = rsqrtf(var + EPSc);
        }
        __syncthreads();
    }
    for (int c = g; c < Hc; c += TPR) {
        float y = ldsY[r * Hc + c];
        if (do_ln)   y = (y - rowM[r]) * rowR[r] * gamma[c] + beta[c];
        if (do_relu) y = fmaxf(y, 0.f);
        out[(row0 + r) * Hc + c] = y;
    }
}

// ---------------------------------------------------------------------------
// Segment mean path (atomic fp32 scatter-add), enrichment, attention.
// ---------------------------------------------------------------------------
__global__ __launch_bounds__(256) void k_segsum(
    const float* __restrict__ P, const int* __restrict__ ids,
    float* __restrict__ sums, float* __restrict__ cnts, i64 ntot)
{
    i64 i = (i64)blockIdx.x * 256 + threadIdx.x;
    if (i >= ntot) return;
    int pt = (int)(i >> 8);
    int c  = (int)(i & 255);
    int id = ids[pt];
    if (id < 0) return;
    int b = pt / NPc;
    atomicAdd(&sums[((i64)(b * Sc + id) << 8) + c], P[i]);
    if (c == 0) atomicAdd(&cnts[b * Sc + id], 1.0f);
}

__global__ __launch_bounds__(256) void k_mean(
    float* __restrict__ sums, const float* __restrict__ cnts, i64 ntot)
{
    i64 i = (i64)blockIdx.x * 256 + threadIdx.x;
    if (i >= ntot) return;
    float c = cnts[i >> 8];
    sums[i] = sums[i] / fmaxf(c, 1.0f);
}

__global__ __launch_bounds__(256) void k_enrich(
    const float* __restrict__ P, float* __restrict__ vb_io,
    const float* __restrict__ sem, const int* __restrict__ ids, i64 ntot)
{
    i64 i = (i64)blockIdx.x * 256 + threadIdx.x;
    if (i >= ntot) return;
    int pt = (int)(i >> 8);
    int c  = (int)(i & 255);
    int id = ids[pt];
    int b  = pt / NPc;
    float p = P[i];
    float s = (id >= 0) ? sem[((i64)(b * Sc + id) << 8) + c] : p;
    vb_io[i] = p + 0.5f * vb_io[i] + 0.5f * s;
}

__global__ __launch_bounds__(256) void k_attn(
    const float* __restrict__ q, const float* __restrict__ Kb,
    const float* __restrict__ Vb, const unsigned char* __restrict__ mask,
    float* __restrict__ out)
{
    __shared__ _Float16 Kl[LTc * Hc];
    __shared__ _Float16 Vl[LTc * Hc];
    __shared__ unsigned char Ml[LTc];

    const int tid = threadIdx.x;
    const int b   = blockIdx.y;
    const i64 tbase = (i64)b * LTc * Hc;
    for (int i = tid; i < LTc * Hc; i += 256) {
        Kl[i] = (_Float16)Kb[tbase + i];
        Vl[i] = (_Float16)Vb[tbase + i];
    }
    if (tid < LTc) Ml[tid] = mask[b * LTc + tid];
    __syncthreads();

    const int h  = tid & 7;
    const int pi = tid >> 3;
    const i64 pt = (i64)b * NPc + (i64)blockIdx.x * 32 + pi;

    float qr[32];
    const float* qp = &q[pt * Hc + h * DHc];
    #pragma unroll
    for (int d = 0; d < 32; ++d) qr[d] = qp[d];

    float m = -3.0e38f, l = 0.f;
    float acc[32];
    #pragma unroll
    for (int d = 0; d < 32; ++d) acc[d] = 0.f;

    for (int j = 0; j < LTc; ++j) {
        float s = 0.f;
        const _Float16* kp = &Kl[j * Hc + h * DHc];
        #pragma unroll
        for (int d = 0; d < 32; ++d) s += qr[d] * (float)kp[d];
        s *= 0.17677669529663687f;            // 1/sqrt(32)
        if (!Ml[j]) s = -1.0e9f;
        float mn   = fmaxf(m, s);
        float corr = __expf(m - mn);
        float w    = __expf(s - mn);
        l = l * corr + w;
        const _Float16* vp = &Vl[j * Hc + h * DHc];
        #pragma unroll
        for (int d = 0; d < 32; ++d) acc[d] = acc[d] * corr + w * (float)vp[d];
        m = mn;
    }
    float inv = 1.f / l;
    float* op = &out[pt * Hc + h * DHc];
    #pragma unroll
    for (int d = 0; d < 32; ++d) op[d] = acc[d] * inv;
}

// ---------------------------------------------------------------------------
// Host orchestration. Workspace: f16 weight pool first (16B-aligned), then
// text/segment buffers, then 3 big 134MB activation buffers (~424 MB total).
// ---------------------------------------------------------------------------
extern "C" void kernel_launch(void* const* d_in, const int* in_sizes, int n_in,
                              void* d_out, int out_size, void* d_ws, size_t ws_size,
                              hipStream_t stream)
{
    (void)in_sizes; (void)n_in; (void)out_size; (void)ws_size;

    const float* pf    = (const float*)d_in[0];
    const float* tf    = (const float*)d_in[1];
    const float* zpv   = (const float*)d_in[2];
    const float* ztv   = (const float*)d_in[3];
    const int*   ids   = (const int*)d_in[4];
    const unsigned char* tmask = (const unsigned char*)d_in[5];

    const float* Wp  = (const float*)d_in[6];  const float* bp  = (const float*)d_in[7];
    const float* gp  = (const float*)d_in[8];  const float* bep = (const float*)d_in[9];
    const float* Wt  = (const float*)d_in[10]; const float* bt  = (const float*)d_in[11];
    const float* gt  = (const float*)d_in[12]; const float* bet = (const float*)d_in[13];
    const float* Wv1 = (const float*)d_in[14]; const float* bv1 = (const float*)d_in[15];
    const float* gv  = (const float*)d_in[16]; const float* bev = (const float*)d_in[17];
    const float* Wv2 = (const float*)d_in[18]; const float* bv2 = (const float*)d_in[19];
    const float* Ws1 = (const float*)d_in[20]; const float* bs1 = (const float*)d_in[21];
    const float* gs  = (const float*)d_in[22]; const float* bes = (const float*)d_in[23];
    const float* Ws2 = (const float*)d_in[24]; const float* bs2 = (const float*)d_in[25];
    const float* Wq  = (const float*)d_in[26]; const float* bq  = (const float*)d_in[27];
    const float* Wk  = (const float*)d_in[28]; const float* bk  = (const float*)d_in[29];
    const float* Wv  = (const float*)d_in[30]; const float* bv  = (const float*)d_in[31];
    const float* Wo  = (const float*)d_in[32]; const float* bo  = (const float*)d_in[33];
    const float* Wi  = (const float*)d_in[34]; const float* bi  = (const float*)d_in[35];
    const float* gi  = (const float*)d_in[36]; const float* bei = (const float*)d_in[37];
    const float* Wout= (const float*)d_in[38]; const float* bout= (const float*)d_in[39];

    const i64 NTXT = (i64)Bc * LTc;     // 512
    const i64 NPTS = (i64)Bc * NPc;     // 131072
    const i64 NSEG = (i64)Bc * Sc;      // 8192

    // f16 weight pool at ws start (16B aligned by allocation).
    _Float16* wp16 = (_Float16*)d_ws;
    i64 h = 0;
    _Float16* WTp  = wp16 + h; h += 256 * Hc;   // 65536
    _Float16* WTt  = wp16 + h; h += 768 * Hc;
    _Float16* WTv1 = wp16 + h; h += 512 * Hc;
    _Float16* WTv2 = wp16 + h; h += 256 * Hc;
    _Float16* WTs1 = wp16 + h; h += 256 * Hc;
    _Float16* WTs2 = wp16 + h; h += 256 * Hc;
    _Float16* WTq  = wp16 + h; h += 256 * Hc;
    _Float16* WTk  = wp16 + h; h += 256 * Hc;
    _Float16* WTv  = wp16 + h; h += 256 * Hc;
    _Float16* WTo  = wp16 + h; h += 256 * Hc;
    _Float16* WTi  = wp16 + h; h += 512 * Hc;
    _Float16* WTout= wp16 + h; h += 256 * Hc;   // h == 1048576 halves

    float* ws = (float*)d_ws + (h + 1) / 2;     // 524288 floats consumed
    i64 o = 0;
    float* T    = ws + o; o += NTXT * Hc;
    float* Kb   = ws + o; o += NTXT * Hc;
    float* Vb   = ws + o; o += NTXT * Hc;
    float* sums = ws + o; o += NSEG * Hc;       // sums -> means -> sem
    float* cnts = ws + o; o += NSEG;
    float* semh = ws + o; o += NSEG * Hc;
    float* b0   = ws + o; o += NPTS * Hc;       // P
    float* b1   = ws + o; o += NPTS * Hc;       // vb -> P_enr -> attnout -> ao
    float* b2   = ws + o; o += NPTS * Hc;       // vb hidden -> q -> zi

    hipMemsetAsync(sums, 0, (size_t)(NSEG * Hc + NSEG) * sizeof(float), stream);

    dim3 blk(256);
    // Weight conversion pre-pass (idempotent, deterministic).
    k_wconv<<<256, blk, 0, stream>>>(Wp,  WTp,  256);
    k_wconv<<<768, blk, 0, stream>>>(Wt,  WTt,  768);
    k_wconv<<<512, blk, 0, stream>>>(Wv1, WTv1, 512);
    k_wconv<<<256, blk, 0, stream>>>(Wv2, WTv2, 256);
    k_wconv<<<256, blk, 0, stream>>>(Ws1, WTs1, 256);
    k_wconv<<<256, blk, 0, stream>>>(Ws2, WTs2, 256);
    k_wconv<<<256, blk, 0, stream>>>(Wq,  WTq,  256);
    k_wconv<<<256, blk, 0, stream>>>(Wk,  WTk,  256);
    k_wconv<<<256, blk, 0, stream>>>(Wv,  WTv,  256);
    k_wconv<<<256, blk, 0, stream>>>(Wo,  WTo,  256);
    k_wconv<<<512, blk, 0, stream>>>(Wi,  WTi,  512);
    k_wconv<<<256, blk, 0, stream>>>(Wout,WTout,256);

    // Text path: T = relu(LN(tf @ Wt + bt)); then K,V projections.
    gemm256<64,768><<<(unsigned)(NTXT / 64), blk, 0, stream>>>(tf, 768, nullptr, 0, WTt, bt, gt, bet, T, 1, 1);
    gemm256<64,256><<<(unsigned)(NTXT / 64), blk, 0, stream>>>(T, 256, nullptr, 0, WTk, bk, nullptr, nullptr, Kb, 0, 0);
    gemm256<64,256><<<(unsigned)(NTXT / 64), blk, 0, stream>>>(T, 256, nullptr, 0, WTv, bv, nullptr, nullptr, Vb, 0, 0);
    // Point path: P = relu(LN(pf @ Wp + bp))
    gemm256<64,256><<<(unsigned)(NPTS / 64), blk, 0, stream>>>(pf, 256, nullptr, 0, WTp, bp, gp, bep, b0, 1, 1);
    // Visual branch: vb = relu(LN([Z_PV|Z_TV] @ Wv1 + bv1)) @ Wv2 + bv2
    gemm256<64,512><<<(unsigned)(NPTS / 64), blk, 0, stream>>>(zpv, 256, ztv, 256, WTv1, bv1, gv, bev, b2, 1, 1);
    gemm256<64,256><<<(unsigned)(NPTS / 64), blk, 0, stream>>>(b2, 256, nullptr, 0, WTv2, bv2, nullptr, nullptr, b1, 0, 0);
    // Superpoint semantics: segment mean -> MLP -> sem (back into `sums`)
    i64 ntot = NPTS * Hc;
    k_segsum<<<(unsigned)((ntot + 255) / 256), blk, 0, stream>>>(b0, ids, sums, cnts, ntot);
    i64 nseg_tot = NSEG * Hc;
    k_mean<<<(unsigned)((nseg_tot + 255) / 256), blk, 0, stream>>>(sums, cnts, nseg_tot);
    gemm256<64,256><<<(unsigned)(NSEG / 64), blk, 0, stream>>>(sums, 256, nullptr, 0, WTs1, bs1, gs, bes, semh, 1, 1);
    gemm256<64,256><<<(unsigned)(NSEG / 64), blk, 0, stream>>>(semh, 256, nullptr, 0, WTs2, bs2, nullptr, nullptr, sums, 0, 0);
    // Enrichment: b1 = P + 0.5*vb + 0.5*P_sem
    k_enrich<<<(unsigned)((ntot + 255) / 256), blk, 0, stream>>>(b0, b1, sums, ids, ntot);
    // q projection, attention, output projection.
    gemm256<64,256><<<(unsigned)(NPTS / 64), blk, 0, stream>>>(b1, 256, nullptr, 0, WTq, bq, nullptr, nullptr, b2, 0, 0);
    k_attn<<<dim3(NPc / 32, Bc), blk, 0, stream>>>(b2, Kb, Vb, tmask, b1);
    gemm256<64,256><<<(unsigned)(NPTS / 64), blk, 0, stream>>>(b1, 256, nullptr, 0, WTo, bo, nullptr, nullptr, b1, 0, 0);
    // zi = relu(LN([P|ao] @ Wi + bi)); out = zi @ Wout + bout
    gemm256<64,512><<<(unsigned)(NPTS / 64), blk, 0, stream>>>(b0, 256, b1, 256, WTi, bi, gi, bei, b2, 1, 1);
    gemm256<64,256><<<(unsigned)(NPTS / 64), blk, 0, stream>>>(b2, 256, nullptr, 0, WTout, bout, nullptr, nullptr,
                                                              (float*)d_out, 0, 0);
}
